// RoPE2DAttention_54176717471897
// MI455X (gfx1250) — matrix operand order repaired and checked by reading, not compiled, not run on hardware
//
#include <hip/hip_runtime.h>
#include <hip/hip_bf16.h>

typedef __attribute__((ext_vector_type(16))) _Float16 v16h;
typedef __attribute__((ext_vector_type(8)))  _Float16 v8h;
typedef __attribute__((ext_vector_type(8)))  float    v8f;

#define BATCH   8
#define SEQ     1024
#define DMODEL  768
#define NHEADS  12
#define HDIM    64
#define L2E     1.44269504088896f

// ---------------------------------------------------------------------------
// WMMA helpers (gfx1250, wave32). Layouts per CDNA5 ISA 7.12.2:
//   A (16x32 f16): lane holds row (lane&15); elems 0..7 = K (lane>>4)*8+e,
//                  elems 8..15 = K 16+(lane>>4)*8+(e-8)
//   B (32x16 f16): lane holds col (lane&15); elems 0..15 use the same K order as A
//   C/D (16x16 f32): lane holds col (lane&15); VGPR v = row v+(lane>>4)*8
// ---------------------------------------------------------------------------
__device__ __forceinline__ v8f wmma_f16(v16h a, v16h b, v8f c) {
  return __builtin_amdgcn_wmma_f32_16x16x32_f16(false, a, false, b,
                                                (short)0, c, false, false);
}

// A-fragment from row-major [M x K] tile, row stride ld (multiple of 8 halfs)
__device__ __forceinline__ v16h load_frag_a(const _Float16* base, int ld) {
  int lane = threadIdx.x & 31;
  const _Float16* p = base + (lane & 15) * ld + ((lane >> 4) << 3);
  v8h lo = *(const v8h*)p;
  v8h hi = *(const v8h*)(p + 16);
  v16h r;
#pragma unroll
  for (int i = 0; i < 8; ++i) { r[i] = lo[i]; r[i + 8] = hi[i]; }
  return r;
}

// B-fragment from "transposed" storage: row = N index, two 8-half K runs 16 apart (same K order as A)
__device__ __forceinline__ v16h load_frag_b(const _Float16* baseT, int ld) {
  int lane = threadIdx.x & 31;
  const _Float16* p = baseT + (lane & 15) * ld + ((lane >> 4) << 3);
  v8h lo = *(const v8h*)p;
  v8h hi = *(const v8h*)(p + 16);
  v16h r;
#pragma unroll
  for (int i = 0; i < 8; ++i) { r[i] = lo[i]; r[i + 8] = hi[i]; }
  return r;
}

// ---------------------------------------------------------------------------
// Ping-pong pipelined 32x64 GEMM tile: C[2][4] += A(32x768) * B(768x64).
// Two register sets alternate producer/consumer roles each half-iteration,
// so no inter-iteration register copies are needed; waits stay partial.
// ---------------------------------------------------------------------------
__device__ __forceinline__ void gemm_loads(const _Float16* __restrict__ A,
                                           const _Float16* __restrict__ Bt,
                                           int kb, v16h a[2], v16h b[4]) {
  a[0] = load_frag_a(A + kb, DMODEL);
  a[1] = load_frag_a(A + 16 * DMODEL + kb, DMODEL);
#pragma unroll
  for (int j = 0; j < 4; ++j)
    b[j] = load_frag_b(Bt + j * 16 * DMODEL + kb, DMODEL);
}

__device__ __forceinline__ void gemm_mmas(const v16h a[2], const v16h b[4],
                                          v8f acc[2][4]) {
#pragma unroll
  for (int j = 0; j < 4; ++j) {
    acc[0][j] = wmma_f16(a[0], b[j], acc[0][j]);
    acc[1][j] = wmma_f16(a[1], b[j], acc[1][j]);
  }
}

__device__ __forceinline__ void gemm_tile_32x64(const _Float16* __restrict__ A,
                                                const _Float16* __restrict__ Bt,
                                                v8f acc[2][4]) {
  v16h aA[2], bA[4], aB[2], bB[4];
  gemm_loads(A, Bt, 0, aA, bA);
  for (int kb = 0; kb < DMODEL; kb += 64) {
    gemm_loads(A, Bt, kb + 32, aB, bB);   // always in range (768 % 64 == 0)
    gemm_mmas(aA, bA, acc);
    if (kb + 64 < DMODEL) gemm_loads(A, Bt, kb + 64, aA, bA);
    gemm_mmas(aB, bB, acc);
  }
}

// ---------------------------------------------------------------------------
// Precision staging: f32 -> f16 cast; weights also transposed to [N][K]
// ---------------------------------------------------------------------------
__global__ void cast_f16_kernel(const float* __restrict__ in,
                                _Float16* __restrict__ out, int n) {
  int i = blockIdx.x * blockDim.x + threadIdx.x;
  if (i < n) out[i] = (_Float16)in[i];
}

// LDS-tiled transpose+cast: coalesced reads and writes. Block 256 = 32x8.
__global__ __launch_bounds__(256) void transpose_f16_kernel(
    const float* __restrict__ w, _Float16* __restrict__ wt) {
  __shared__ float tile[32][33];
  const int kb = blockIdx.x * 32;  // source row block (K)
  const int nb = blockIdx.y * 32;  // source col block (N)
  const int tx = threadIdx.x & 31, ty = threadIdx.x >> 5;
#pragma unroll
  for (int i = 0; i < 4; ++i)
    tile[ty + i * 8][tx] = w[(size_t)(kb + ty + i * 8) * DMODEL + nb + tx];
  __syncthreads();
#pragma unroll
  for (int i = 0; i < 4; ++i)
    wt[(size_t)(nb + ty + i * 8) * DMODEL + kb + tx] =
        (_Float16)tile[tx][ty + i * 8];
}

// ---------------------------------------------------------------------------
// QKV projection: one wave per 32x64 tile of [8192 x 768] output.
// Fused bias + 2D RoPE (q,k) and transposed store for v.
// blockIdx.z: 0=Q, 1=K, 2=V
// ---------------------------------------------------------------------------
__global__ __launch_bounds__(128) void qkv_rope_kernel(
    const _Float16* __restrict__ xh,
    const _Float16* __restrict__ wtq, const _Float16* __restrict__ wtk,
    const _Float16* __restrict__ wtv,
    const float* __restrict__ bq, const float* __restrict__ bk,
    const float* __restrict__ bv,
    _Float16* __restrict__ qbuf, _Float16* __restrict__ kbuf,
    _Float16* __restrict__ vtbuf) {
  const int mode = blockIdx.z;
  const _Float16* wt = (mode == 0) ? wtq : ((mode == 1) ? wtk : wtv);
  const float* bias = (mode == 0) ? bq : ((mode == 1) ? bk : bv);

  const int wid = blockIdx.x * 4 + (threadIdx.x >> 5);  // 256*12 tiles
  const int mt = wid / 12, nt = wid % 12;
  const int m0 = mt * 32, n0 = nt * 64;
  const int lane = threadIdx.x & 31, hiL = lane >> 4, col = lane & 15;

  v8f acc[2][4];
#pragma unroll
  for (int t = 0; t < 2; ++t)
#pragma unroll
    for (int j = 0; j < 4; ++j)
#pragma unroll
      for (int r = 0; r < 8; ++r) acc[t][j][r] = 0.0f;

  gemm_tile_32x64(xh + (size_t)m0 * DMODEL, wt + (size_t)n0 * DMODEL, acc);

#pragma unroll
  for (int j = 0; j < 4; ++j) {
    const int d = n0 + j * 16 + col;
    const float bias_d = bias[d];
    const int h = d >> 6, hd = d & 63;
    // RoPE pair index i = (hd mod 32)/2; freq = 10000^(-i/16)
    const int pi = (hd & 31) >> 1;
    const float fr = exp2f(-(float)pi * 0.8304820237218406f); // log2(1e4)/16
#pragma unroll
    for (int t = 0; t < 2; ++t) {
#pragma unroll
      for (int r = 0; r < 8; ++r) {
        const int m = m0 + t * 16 + r + hiL * 8;
        const int bI = m >> 10, s = m & 1023;
        float val = acc[t][j][r] + bias_d;
        if (mode == 2) {
          // v stored transposed: [b][h][hd][s]
          vtbuf[(((size_t)bI * NHEADS + h) * HDIM + hd) * SEQ + s] =
              (_Float16)val;
        } else {
          // 2D RoPE: dims<32 rotate with row=s/32, dims>=32 with col=s%32
          float partner = __shfl_xor(val, 1, 32);  // adjacent hd/lane pair
          float pos = (hd < 32) ? (float)(s >> 5) : (float)(s & 31);
          float sn, cs;
          __sincosf(pos * fr, &sn, &cs);
          float res = ((hd & 1) == 0) ? (val * cs - partner * sn)
                                      : (partner * sn + val * cs);
          if (mode == 0) res *= 0.125f;  // fold 1/sqrt(64) into q
          _Float16* outp = (mode == 0) ? qbuf : kbuf;
          outp[(((size_t)bI * NHEADS + h) * SEQ + s) * HDIM + hd] =
              (_Float16)res;
        }
      }
    }
  }
}

// ---------------------------------------------------------------------------
// Flash attention: one wave per 16 queries, streaming 32 keys / step.
// One 32-key block: V frags for this block and K frags for the next block
// are issued before the score WMMAs; softmax VALU overlaps in-flight VMEM.
// kc = current K frags (read), kn = next K frags (written if prefetch).
// ---------------------------------------------------------------------------
__device__ __forceinline__ void attn_block(
    int kt, int ktn, bool prefetch, const _Float16* __restrict__ kp,
    const _Float16* __restrict__ vp, v16h aq0, v16h aq1, v16h kc[4],
    v16h kn[4], v8f acc[4], float mrow[8], float lrow[8], _Float16* ps,
    int hiL, int col) {
  v16h vf[4];
#pragma unroll
  for (int j = 0; j < 4; ++j)
    vf[j] = load_frag_b(vp + (size_t)(j * 16) * SEQ + kt, SEQ);
  if (prefetch) {
    const _Float16* kpn = kp + (size_t)ktn * HDIM;
    kn[0] = load_frag_b(kpn, HDIM);
    kn[1] = load_frag_b(kpn + 32, HDIM);
    kn[2] = load_frag_b(kpn + 16 * HDIM, HDIM);
    kn[3] = load_frag_b(kpn + 16 * HDIM + 32, HDIM);
  }

  // scores for 16 queries x 32 keys (q was pre-scaled by 1/8)
  v8f s0, s1;
#pragma unroll
  for (int r = 0; r < 8; ++r) { s0[r] = 0.0f; s1[r] = 0.0f; }
  s0 = wmma_f16(aq0, kc[0], s0);
  s0 = wmma_f16(aq1, kc[1], s0);
  s1 = wmma_f16(aq0, kc[2], s1);
  s1 = wmma_f16(aq1, kc[3], s1);

  // online softmax update (row = query, spread over 16 lanes)
#pragma unroll
  for (int r = 0; r < 8; ++r) {
    float mx = fmaxf(s0[r], s1[r]);
#pragma unroll
    for (int off = 1; off < 16; off <<= 1)
      mx = fmaxf(mx, __shfl_xor(mx, off, 32));
    float mnew = fmaxf(mrow[r], mx);
    float alpha = exp2f((mrow[r] - mnew) * L2E);
    float p0 = exp2f((s0[r] - mnew) * L2E);
    float p1 = exp2f((s1[r] - mnew) * L2E);
    float rs = p0 + p1;
#pragma unroll
    for (int off = 1; off < 16; off <<= 1) rs += __shfl_xor(rs, off, 32);
    lrow[r] = lrow[r] * alpha + rs;
    mrow[r] = mnew;
    acc[0][r] *= alpha; acc[1][r] *= alpha;
    acc[2][r] *= alpha; acc[3][r] *= alpha;
    const int prow = r + hiL * 8;  // stage probs (C/D layout) into LDS
    ps[prow * 32 + col]      = (_Float16)p0;
    ps[prow * 32 + 16 + col] = (_Float16)p1;
  }

  // re-read probabilities in A-fragment layout (per-wave LDS, in-order DS)
  asm volatile("s_wait_dscnt 0x0" ::: "memory");
  v16h pa;
  {
    const _Float16* pr = ps + col * 32 + hiL * 8;
    v8h lo = *(const v8h*)pr;
    v8h hi = *(const v8h*)(pr + 16);
#pragma unroll
    for (int i = 0; i < 8; ++i) { pa[i] = lo[i]; pa[i + 8] = hi[i]; }
  }

  // PV: 16x32 probs x 32x64 values (vT gives natural B layout)
#pragma unroll
  for (int j = 0; j < 4; ++j) acc[j] = wmma_f16(pa, vf[j], acc[j]);
}

__global__ __launch_bounds__(128) void attn_kernel(
    const _Float16* __restrict__ qbuf, const _Float16* __restrict__ kbuf,
    const _Float16* __restrict__ vtbuf, _Float16* __restrict__ obuf) {
  __shared__ __align__(16) _Float16 pstage[4 * 16 * 32];  // 4 waves x 16x32

  const int wavel = threadIdx.x >> 5;
  const int wid = blockIdx.x * 4 + wavel;
  const int bh = wid >> 6;           // 64 query tiles per (b,h)
  const int q0 = (wid & 63) << 4;
  const int lane = threadIdx.x & 31, hiL = lane >> 4, col = lane & 15;
  const int bI = bh / NHEADS, h = bh % NHEADS;

  const _Float16* qp = qbuf + ((size_t)bh * SEQ + q0) * HDIM;
  const _Float16* kp = kbuf + (size_t)bh * SEQ * HDIM;
  const _Float16* vp = vtbuf + (size_t)bh * HDIM * SEQ;
  _Float16* ps = &pstage[wavel * 512];

  // q A-fragments (16 queries x 64 head dim), hoisted out of the key loop
  const v16h aq0 = load_frag_a(qp, HDIM);
  const v16h aq1 = load_frag_a(qp + 32, HDIM);

  v8f acc[4];
  float mrow[8], lrow[8];
#pragma unroll
  for (int r = 0; r < 8; ++r) { mrow[r] = -1.0e30f; lrow[r] = 0.0f; }
#pragma unroll
  for (int j = 0; j < 4; ++j)
#pragma unroll
    for (int r = 0; r < 8; ++r) acc[j][r] = 0.0f;

  // ping-pong K fragment sets: no copies between iterations
  v16h kcA[4], kcB[4];
  kcA[0] = load_frag_b(kp, HDIM);
  kcA[1] = load_frag_b(kp + 32, HDIM);
  kcA[2] = load_frag_b(kp + 16 * HDIM, HDIM);
  kcA[3] = load_frag_b(kp + 16 * HDIM + 32, HDIM);

  for (int kt = 0; kt < SEQ; kt += 64) {
    // phase 0: compute on set A, prefetch set B @ kt+32 (always in range)
    attn_block(kt, kt + 32, true, kp, vp, aq0, aq1, kcA, kcB, acc, mrow,
               lrow, ps, hiL, col);
    // phase 1: compute on set B, prefetch set A @ kt+64 (guarded)
    attn_block(kt + 32, kt + 64, (kt + 64) < SEQ, kp, vp, aq0, aq1, kcB,
               kcA, acc, mrow, lrow, ps, hiL, col);
  }

  // normalize and store merged-head output [b][s][h*64+hd] as f16
#pragma unroll
  for (int j = 0; j < 4; ++j)
#pragma unroll
    for (int r = 0; r < 8; ++r) {
      float o = acc[j][r] / lrow[r];
      int s = q0 + r + hiL * 8;
      obuf[((size_t)bI * SEQ + s) * DMODEL + h * HDIM + j * 16 + col] =
          (_Float16)o;
    }
}

// ---------------------------------------------------------------------------
// Output projection: [8192 x 768] f16 @ Wo^T f16 + bo -> f32, 32x64 tiles
// ---------------------------------------------------------------------------
__global__ __launch_bounds__(128) void out_proj_kernel(
    const _Float16* __restrict__ ah, const _Float16* __restrict__ wto,
    const float* __restrict__ bo, float* __restrict__ out) {
  const int wid = blockIdx.x * 4 + (threadIdx.x >> 5);
  const int mt = wid / 12, nt = wid % 12;
  const int m0 = mt * 32, n0 = nt * 64;
  const int lane = threadIdx.x & 31, hiL = lane >> 4, col = lane & 15;

  v8f acc[2][4];
#pragma unroll
  for (int t = 0; t < 2; ++t)
#pragma unroll
    for (int j = 0; j < 4; ++j)
#pragma unroll
      for (int r = 0; r < 8; ++r) acc[t][j][r] = 0.0f;

  gemm_tile_32x64(ah + (size_t)m0 * DMODEL, wto + (size_t)n0 * DMODEL, acc);

#pragma unroll
  for (int j = 0; j < 4; ++j) {
    const int d = n0 + j * 16 + col;
    const float bd = bo[d];
#pragma unroll
    for (int t = 0; t < 2; ++t)
#pragma unroll
      for (int r = 0; r < 8; ++r) {
        const int m = m0 + t * 16 + r + hiL * 8;
        out[(size_t)m * DMODEL + d] = acc[t][j][r] + bd;
      }
  }
}

// ---------------------------------------------------------------------------
extern "C" void kernel_launch(void* const* d_in, const int* in_sizes, int n_in,
                              void* d_out, int out_size, void* d_ws,
                              size_t ws_size, hipStream_t stream) {
  const float* x  = (const float*)d_in[0];
  const float* Wq = (const float*)d_in[1];
  const float* bq = (const float*)d_in[2];
  const float* Wk = (const float*)d_in[3];
  const float* bk = (const float*)d_in[4];
  const float* Wv = (const float*)d_in[5];
  const float* bv = (const float*)d_in[6];
  const float* Wo = (const float*)d_in[7];
  const float* bo = (const float*)d_in[8];
  float* out = (float*)d_out;

  char* ws = (char*)d_ws;
  size_t off = 0;
  auto alloc = [&](size_t bytes) -> void* {
    void* p = ws + off;
    off += (bytes + 255) & ~(size_t)255;
    return p;
  };
  const size_t actsz = (size_t)BATCH * SEQ * DMODEL * sizeof(_Float16);
  const size_t wsz   = (size_t)DMODEL * DMODEL * sizeof(_Float16);
  _Float16* xh    = (_Float16*)alloc(actsz);
  _Float16* wtq   = (_Float16*)alloc(wsz);
  _Float16* wtk   = (_Float16*)alloc(wsz);
  _Float16* wtv   = (_Float16*)alloc(wsz);
  _Float16* wto   = (_Float16*)alloc(wsz);
  _Float16* qbuf  = (_Float16*)alloc(actsz);
  _Float16* kbuf  = (_Float16*)alloc(actsz);
  _Float16* vtbuf = (_Float16*)alloc(actsz);
  _Float16* obuf  = (_Float16*)alloc(actsz);
  (void)ws_size; (void)in_sizes; (void)n_in; (void)out_size;

  const int nX = BATCH * SEQ * DMODEL;
  cast_f16_kernel<<<(nX + 255) / 256, 256, 0, stream>>>(x, xh, nX);
  dim3 tgrid(DMODEL / 32, DMODEL / 32);
  transpose_f16_kernel<<<tgrid, 256, 0, stream>>>(Wq, wtq);
  transpose_f16_kernel<<<tgrid, 256, 0, stream>>>(Wk, wtk);
  transpose_f16_kernel<<<tgrid, 256, 0, stream>>>(Wv, wtv);
  transpose_f16_kernel<<<tgrid, 256, 0, stream>>>(Wo, wto);

  // 256 m-tiles x 12 n-tiles = 3072 waves; 4 waves/block -> 768 blocks
  dim3 gqkv(768, 1, 3);
  qkv_rope_kernel<<<gqkv, 128, 0, stream>>>(xh, wtq, wtk, wtv, bq, bk, bv,
                                            qbuf, kbuf, vtbuf);
  // 8*12*64 = 6144 waves -> 1536 blocks
  attn_kernel<<<1536, 128, 0, stream>>>(qbuf, kbuf, vtbuf, obuf);
  out_proj_kernel<<<768, 128, 0, stream>>>(obuf, wto, bo, out);
}
